// MultiHeadCrossAttention_70437463654802
// MI455X (gfx1250) — matrix operand order repaired and checked by reading, not compiled, not run on hardware
//
#include <hip/hip_runtime.h>

// ---------------- constants ----------------
#define TDIM 512        // IMG == T == A
#define NH   8
#define HD   64
#define NBC  24         // B*C
#define NBH  192        // B*C*H

typedef __attribute__((ext_vector_type(16))) __bf16        v16bf;
typedef __attribute__((ext_vector_type(8)))  __bf16        v8bf;
typedef __attribute__((ext_vector_type(8)))  float         v8f;
typedef __attribute__((ext_vector_type(8)))  unsigned short us8;
typedef __attribute__((ext_vector_type(4)))  int           v4i;

// ---- CDNA5 async global->LDS copy (probe with __has_builtin, safe fallback) ----
#if defined(__has_builtin)
#if __has_builtin(__builtin_amdgcn_global_load_async_to_lds_b128)
#define HAS_ASYNC_LDS 1
#endif
#if __has_builtin(__builtin_amdgcn_s_wait_asynccnt)
#define HAS_WAIT_ASYNC_BUILTIN 1
#endif
#endif
#ifndef HAS_ASYNC_LDS
#define HAS_ASYNC_LDS 0
#endif
#ifndef HAS_WAIT_ASYNC_BUILTIN
#define HAS_WAIT_ASYNC_BUILTIN 0
#endif

typedef __attribute__((address_space(1))) v4i* gas_v4i;   // global int4*
typedef __attribute__((address_space(3))) v4i* las_v4i;   // LDS int4*

__device__ __forceinline__ void async_cp16(const unsigned short* g, unsigned short* l) {
#if HAS_ASYNC_LDS
  __builtin_amdgcn_global_load_async_to_lds_b128((gas_v4i)g, (las_v4i)l, 0, 0);
#else
  *(us8*)l = *(const us8*)g;
#endif
}

__device__ __forceinline__ void wait_async0() {
#if HAS_ASYNC_LDS
#if HAS_WAIT_ASYNC_BUILTIN
  __builtin_amdgcn_s_wait_asynccnt(0);
#else
  asm volatile("s_wait_asynccnt 0x0" ::: "memory");
#endif
#endif
}

__device__ __forceinline__ unsigned short f2bf(float f) {
  union { float f; unsigned u; } v; v.f = f;
  unsigned u = v.u;
  u += 0x7FFFu + ((u >> 16) & 1u);          // round-to-nearest-even
  return (unsigned short)(u >> 16);
}

// ---------------- elementwise kernels ----------------
__global__ __launch_bounds__(256) void cvt_f32_bf16_k(const float* __restrict__ s,
                                                      unsigned short* __restrict__ d, int n) {
  int i = blockIdx.x * 256 + threadIdx.x;
  if (i < n) d[i] = f2bf(s[i]);
}

__global__ __launch_bounds__(256) void add3_bf16_k(const float* __restrict__ a,
                                                   const float* __restrict__ b,
                                                   const float* __restrict__ c,
                                                   unsigned short* __restrict__ d, int n) {
  int i = blockIdx.x * 256 + threadIdx.x;
  if (i < n) d[i] = f2bf(a[i] + b[i] + c[i]);
}

// ---------------- generic batched WMMA GEMM ----------------
// C[z] = alpha * A[z] * B[z] + bias[z]
//   A: bf16 [M,K] row-major, batch index (z / aDiv), stride aStride (elems)
//   B: bf16; TRB=0: [K,N] row-major; TRB=1: [N,K] row-major
//      batch index (z / bDiv) % bMod, stride bStride
//   bias: f32 vector length N (optional), offset (z % biasMod)*biasStride
//   C: f32 or bf16, elem offset = (z/cDiv1)*cStride1 + (z%cMod2)*cStride2 + row*ldc + col
struct GemmP {
  const unsigned short* A; long long aStride; int aDiv;
  const unsigned short* B; long long bStride; int bDiv; int bMod;
  const float* bias; int biasMod; int biasStride;
  void* C; int cIsF32; long long cStride1; int cDiv1; long long cStride2; int cMod2; int ldc;
  int M, N, K; float alpha;
};

template <int BN, int TRB>
__global__ __launch_bounds__(256) void gemm_bf16_k(GemmP p) {
  constexpr int BM = 128, BK = 32, LDP = 40;        // LDP: padded LDS row (elems)
  constexpr int NT = 4;                             // N-tiles of 16 per wave
  constexpr int MT = (BN == 128) ? 2 : 1;           // M-tiles per wave
  constexpr int NEL = (BK * BN) / 256;              // sync B elems per thread
  constexpr int BSTEP = 256 / BN;                   // k stride in sync B staging

  __shared__ unsigned short sA[2][BM * LDP];
  __shared__ unsigned short sB[2][BN * LDP];

  const int tid  = threadIdx.x;
  const int wave = tid >> 5;
  const int lane = tid & 31;
  const int z    = blockIdx.z;
  const int blockM = blockIdx.y * BM;
  const int blockN = blockIdx.x * BN;
  const int K = p.K, N = p.N;

  const unsigned short* Ap =
      p.A + (long long)(z / p.aDiv) * p.aStride + (long long)blockM * K;
  const unsigned short* Bp =
      p.B + (long long)((z / p.bDiv) % p.bMod) * p.bStride;

  const int wM = (BN == 128) ? (wave >> 1) * (MT * 16) : wave * 16;
  const int wN = (BN == 128) ? (wave & 1) * 64 : 0;

  // per-thread staging coordinates: 2x b128 chunks for A (and B when TRB)
  const int ar  = tid >> 2;           // 0..63
  const int akc = (tid & 3) * 8;      // 0,8,16,24

  auto stageA = [&](int k0, int buf) {
    async_cp16(&Ap[(long long)ar * K + k0 + akc],        &sA[buf][ar * LDP + akc]);
    async_cp16(&Ap[(long long)(ar + 64) * K + k0 + akc], &sA[buf][(ar + 64) * LDP + akc]);
  };
  auto stageB = [&](int k0, int buf) {
    if constexpr (TRB) {              // B is [N,K]: vectorized async copy
      async_cp16(&Bp[(long long)(blockN + ar) * K + k0 + akc],
                 &sB[buf][ar * LDP + akc]);
      async_cp16(&Bp[(long long)(blockN + ar + 64) * K + k0 + akc],
                 &sB[buf][(ar + 64) * LDP + akc]);
    } else {                          // B is [K,N]: transpose into [N][K] slots
      const int n  = tid & (BN - 1);
      const int kb = tid / BN;
      unsigned short tmp[NEL];
#pragma unroll
      for (int i = 0; i < NEL; ++i)
        tmp[i] = Bp[(long long)(k0 + kb + i * BSTEP) * N + blockN + n];
      if (k0 + BK < K)
        __builtin_prefetch(&Bp[(long long)(k0 + BK + kb) * N + blockN + n], 0, 1);
#pragma unroll
      for (int i = 0; i < NEL; ++i)
        sB[buf][n * LDP + kb + i * BSTEP] = tmp[i];
    }
  };

  const v8f zacc = {0.f, 0.f, 0.f, 0.f, 0.f, 0.f, 0.f, 0.f};
  v8f acc[MT][NT];
#pragma unroll
  for (int i = 0; i < MT; ++i)
#pragma unroll
    for (int j = 0; j < NT; ++j) acc[i][j] = zacc;

  const int fr  = lane & 15;           // fragment row (A) / col (B)
  const int kbA = (lane >> 4) * 8;     // A frag: chunks at kbA and kbA+16
  const int kbB = (lane >> 4) * 8;     // B frag: chunks at kbB and kbB+16

  // prologue: stage tile 0 into buffer 0
  stageA(0, 0);
  stageB(0, 0);
  wait_async0();
  __syncthreads();

  const int nTiles = K / BK;
  for (int t = 0; t < nTiles; ++t) {
    const int buf = t & 1;
    const int k0  = t * BK;
    if (t + 1 < nTiles) {              // issue next tile into other buffer
      stageA(k0 + BK, buf ^ 1);
      stageB(k0 + BK, buf ^ 1);
    }

    // -------- fragments --------
    v16bf aF[MT], bF[NT];
#pragma unroll
    for (int mt = 0; mt < MT; ++mt) {
      const unsigned short* rp = &sA[buf][(wM + mt * 16 + fr) * LDP];
      v8bf lo = *(const v8bf*)(rp + kbA);
      v8bf hi = *(const v8bf*)(rp + kbA + 16);
#pragma unroll
      for (int e = 0; e < 8; ++e) { aF[mt][e] = lo[e]; aF[mt][e + 8] = hi[e]; }
    }
#pragma unroll
    for (int nt = 0; nt < NT; ++nt) {
      const unsigned short* rp = &sB[buf][(wN + nt * 16 + fr) * LDP + kbB];
      v8bf lo = *(const v8bf*)(rp);
      v8bf hi = *(const v8bf*)(rp + 16);
#pragma unroll
      for (int e = 0; e < 8; ++e) { bF[nt][e] = lo[e]; bF[nt][e + 8] = hi[e]; }
    }
    // -------- MT*NT (8 if BN=128, else 4) WMMA per K-step per wave --------
#pragma unroll
    for (int mt = 0; mt < MT; ++mt)
#pragma unroll
      for (int nt = 0; nt < NT; ++nt)
        acc[mt][nt] = __builtin_amdgcn_wmma_f32_16x16x32_bf16(
            false, aF[mt], false, bF[nt], (short)0, acc[mt][nt], false, false);

    wait_async0();                     // next-tile asyncs land behind the WMMAs
    __syncthreads();
  }

  // -------- epilogue: alpha * acc + bias --------
  const long long cBase = (long long)(z / p.cDiv1) * p.cStride1 +
                          (long long)(z % p.cMod2) * p.cStride2;
  const int rOff    = (lane >> 4) * 8;
  const int biasOff = p.bias ? (z % p.biasMod) * p.biasStride : 0;
#pragma unroll
  for (int mt = 0; mt < MT; ++mt) {
#pragma unroll
    for (int nt = 0; nt < NT; ++nt) {
      int   col = blockN + wN + nt * 16 + fr;
      float bv  = p.bias ? p.bias[biasOff + col] : 0.0f;
#pragma unroll
      for (int r = 0; r < 8; ++r) {
        int       row = blockM + wM + mt * 16 + rOff + r;
        float     vv  = acc[mt][nt][r] * p.alpha + bv;
        long long idx = cBase + (long long)row * p.ldc + col;
        if (p.cIsF32) ((float*)p.C)[idx] = vv;
        else          ((unsigned short*)p.C)[idx] = f2bf(vv);
      }
    }
  }
}

// ---------------- causal softmax (one wave32 per 512-wide row) ----------------
__global__ __launch_bounds__(256) void softmax_causal_k(const float* __restrict__ S,
                                                        unsigned short* __restrict__ P) {
  int row_id = blockIdx.x * 8 + (threadIdx.x >> 5);
  int lane   = threadIdx.x & 31;
  int t      = row_id & (TDIM - 1);
  const float* row = S + (long long)row_id * TDIM;

  float v[16];
  float mx = -__builtin_inff();
#pragma unroll
  for (int i = 0; i < 16; ++i) {
    int   c = lane + i * 32;
    float x = row[c];
    v[i] = (c <= t) ? x : -__builtin_inff();
    mx   = fmaxf(mx, v[i]);
  }
#pragma unroll
  for (int o = 16; o > 0; o >>= 1) mx = fmaxf(mx, __shfl_xor(mx, o, 32));

  float e[16], sum = 0.f;
#pragma unroll
  for (int i = 0; i < 16; ++i) {
    float ex = (v[i] == -__builtin_inff()) ? 0.f : __expf(v[i] - mx);
    e[i] = ex; sum += ex;
  }
#pragma unroll
  for (int o = 16; o > 0; o >>= 1) sum += __shfl_xor(sum, o, 32);
  float inv = 1.0f / sum;

  unsigned short* prow = P + (long long)row_id * TDIM;
#pragma unroll
  for (int i = 0; i < 16; ++i) prow[lane + i * 32] = f2bf(e[i] * inv);
}

// ---------------- host orchestration ----------------
extern "C" void kernel_launch(void* const* d_in, const int* in_sizes, int n_in,
                              void* d_out, int out_size, void* d_ws, size_t ws_size,
                              hipStream_t stream) {
  const float* idx_i  = (const float*)d_in[0];
  const float* idx_t  = (const float*)d_in[1];
  const float* idx_ir = (const float*)d_in[2];
  const float* bli = (const float*)d_in[4];
  const float* blt = (const float*)d_in[6];
  const float* blir = (const float*)d_in[8];
  const float* blo = (const float*)d_in[10];
  const float* bp  = (const float*)d_in[12];
  const float* bk  = (const float*)d_in[14];
  const float* bpi = (const float*)d_in[16];
  const float* bki = (const float*)d_in[18];
  const float* bv  = (const float*)d_in[20];

  const long long TT = (long long)TDIM * TDIM;   // 262144
  const long long TD = (long long)TDIM * HD;     // 32768
  const long long E  = (long long)NBC * TT;      // 6,291,456 (== NBH*TD)
  const long long Es = (long long)NBH * TT;      // 50,331,648

  char* ws = (char*)d_ws;
  auto al = [](long long x) { return (x + 255) & ~255LL; };

  // --- workspace layout (scores-f32 region aliases the dead R0..R8 buffers) ---
  long long off = 0;
  float* wei2 = (float*)(ws);                        // Es * 4 bytes
  unsigned short* R[9];                              // nine [24,512,512] bf16, alias wei2
  for (int i = 0; i < 9; ++i) R[i] = (unsigned short*)(ws + (long long)i * al(E * 2));
  off += al(Es * 4);
  unsigned short* pos  = (unsigned short*)(ws + off); off += al(E * 2);
  unsigned short* keyp = (unsigned short*)(ws + off); off += al(E * 2);
  unsigned short* keyi = (unsigned short*)(ws + off); off += al(E * 2);
  unsigned short* valp = (unsigned short*)(ws + off); off += al(E * 2);
  unsigned short* wei1 = (unsigned short*)(ws + off); off += al(Es * 2);
  unsigned short* prob = wei1;                       // alias (wei1 dead by then)
  unsigned short* wmid = (unsigned short*)(ws + off); off += al(E * 2);
  unsigned short* W[9];
  for (int i = 0; i < 9; ++i) { W[i] = (unsigned short*)(ws + off); off += al((long long)TT * 2); }

  dim3 blk(256);
  int nE = (int)E, nW = (int)TT;

  // --- precision conversion: activations + weights -> bf16 ---
  add3_bf16_k<<<(nE + 255) / 256, blk, 0, stream>>>(idx_i, idx_t, idx_ir, R[0], nE);
  cvt_f32_bf16_k<<<(nE + 255) / 256, blk, 0, stream>>>(idx_t,  R[1], nE);
  cvt_f32_bf16_k<<<(nE + 255) / 256, blk, 0, stream>>>(idx_ir, R[2], nE);
  const float* wsrc[9] = {(const float*)d_in[3],  (const float*)d_in[5],
                          (const float*)d_in[7],  (const float*)d_in[9],
                          (const float*)d_in[11], (const float*)d_in[13],
                          (const float*)d_in[15], (const float*)d_in[17],
                          (const float*)d_in[19]};  // Wli Wlt Wlir Wlo Wp Wk Wpi Wki Wv
  for (int i = 0; i < 9; ++i)
    cvt_f32_bf16_k<<<(nW + 255) / 256, blk, 0, stream>>>(wsrc[i], W[i], nW);

  auto gemm = [&](int BN, const unsigned short* A, long long aS, int aDiv,
                  const unsigned short* Bm, long long bS, int bDiv, int bMod, int trB,
                  const float* bias, int bMd, int bSt,
                  void* Cp, int cF32, long long cS1, int cD1, long long cS2, int cM2, int ldc,
                  int M, int N, int K, float alpha, int NB) {
    GemmP p{A, aS, aDiv, Bm, bS, bDiv, bMod, bias, bMd, bSt,
            Cp, cF32, cS1, cD1, cS2, cM2, ldc, M, N, K, alpha};
    dim3 g(N / BN, M / 128, NB);
    if (BN == 128) {
      if (trB) gemm_bf16_k<128, 1><<<g, blk, 0, stream>>>(p);
      else     gemm_bf16_k<128, 0><<<g, blk, 0, stream>>>(p);
    } else {
      gemm_bf16_k<64, 0><<<g, blk, 0, stream>>>(p);
    }
  };

  const float scale = 0.125f;  // HEAD ** -0.5

  // --- shared linear chain ---
  gemm(128, R[0], TT, 1, W[0], 0, 1, 1, 0, bli, 1, 0, R[3], 0, TT, 1, 0, 1, TDIM, 512, 512, 512, 1.f, NBC); // xi1
  gemm(128, R[1], TT, 1, W[1], 0, 1, 1, 0, blt, 1, 0, R[4], 0, TT, 1, 0, 1, TDIM, 512, 512, 512, 1.f, NBC); // xt
  gemm(128, R[2], TT, 1, W[2], 0, 1, 1, 0, blir,1, 0, R[5], 0, TT, 1, 0, 1, TDIM, 512, 512, 512, 1.f, NBC); // xir
  gemm(128, R[3], TT, 1, R[4], TT, 1, NBC, 0, nullptr, 1, 0, R[6], 0, TT, 1, 0, 1, TDIM, 512, 512, 512, 1.f, NBC); // xi1@xt
  gemm(128, R[6], TT, 1, R[5], TT, 1, NBC, 0, nullptr, 1, 0, R[7], 0, TT, 1, 0, 1, TDIM, 512, 512, 512, 1.f, NBC); // @xir
  gemm(128, R[7], TT, 1, W[3], 0, 1, 1, 0, blo, 1, 0, R[8], 0, TT, 1, 0, 1, TDIM, 512, 512, 512, 1.f, NBC);        // @Wlo -> xi

  // --- per-head projections (z = bc*8 + h) ---
  gemm(64, R[8], TT, 8, W[4], TD, 1, NH, 0, bp,  NH, HD, pos,  0, TD, 1, 0, 1, HD, 512, 64, 512, 1.f, NBH);
  gemm(64, R[4], TT, 8, W[5], TD, 1, NH, 0, bk,  NH, HD, keyp, 0, TD, 1, 0, 1, HD, 512, 64, 512, 1.f, NBH);
  gemm(64, R[5], TT, 8, W[7], TD, 1, NH, 0, bki, NH, HD, keyi, 0, TD, 1, 0, 1, HD, 512, 64, 512, 1.f, NBH);
  gemm(64, R[8], TT, 8, W[8], TD, 1, NH, 0, bv,  NH, HD, valp, 0, TD, 1, 0, 1, HD, 512, 64, 512, 1.f, NBH);

  // --- scores stage 1: pos @ key^T * scale ---
  gemm(128, pos, TD, 1, keyp, TD, 1, NBH, 1, nullptr, 1, 0, wei1, 0, TT, 1, 0, 1, TDIM, 512, 512, 64, scale, NBH);
  // --- mid linear: wei1 @ Wpi[h] + bpi[h] ---
  gemm(64, wei1, TT, 1, W[6], TD, 1, NH, 0, bpi, NH, HD, wmid, 0, TD, 1, 0, 1, HD, 512, 64, 512, 1.f, NBH);
  // --- scores stage 2: wmid @ keyi^T * scale (f32 out, aliases dead R buffers) ---
  gemm(128, wmid, TD, 1, keyi, TD, 1, NBH, 1, nullptr, 1, 0, wei2, 1, TT, 1, 0, 1, TDIM, 512, 512, 64, scale, NBH);

  // --- causal softmax -> bf16 probabilities ---
  softmax_causal_k<<<(NBH * TDIM) / 8, blk, 0, stream>>>(wei2, prob);

  // --- out = prob @ val, head-interleaved f32 store into d_out ---
  gemm(64, prob, TT, 1, valp, TD, 1, NBH, 0, nullptr, 1, 0, d_out, 1,
       TT, NH, HD, NH, TDIM, 512, 64, 512, 1.f, NBH);
}
